// TinySAModule_39496519254440
// MI455X (gfx1250) — matrix operand order repaired and checked by reading, not compiled, not run on hardware
//
#include <hip/hip_runtime.h>
#include <hip/hip_bf16.h>
#include <math.h>

#define NPTS 16384
#define MCTR 4096
#define DIM  128
#define KNB  32
#define HDIM 32
#define RADIUS2 (0.3f*0.3f)
#define ATT_SCALE 0.17677669529663687f
#define LN_EPS 1e-5f

#define AS  136   // A tile LDS stride (halves): 272B rows, bank-rotating, 16B aligned
#define KVS 257   // KV tile LDS stride (floats): 257 % 64 == 1 -> conflict-free column walks

typedef __attribute__((ext_vector_type(16))) _Float16 v16h;
typedef __attribute__((ext_vector_type(8)))  _Float16 v8h;
typedef __attribute__((ext_vector_type(8)))  float    v8f;

// 16-byte transfer unit type for the async global->LDS builtin (matches its parameter type)
typedef int b128_t __attribute__((vector_size(16)));

#if defined(__gfx1250__) && __has_builtin(__builtin_amdgcn_global_load_async_to_lds_b128)
#define USE_ASYNC_LDS 1
#define ASYNC_GPTR(p) ((__attribute__((address_space(1))) b128_t*)(p))
#define ASYNC_SPTR(p) ((__attribute__((address_space(3))) b128_t*)(p))
#else
#define USE_ASYNC_LDS 0
#endif

// ---------------- prep kernels ----------------

__global__ void k_f2h(const float* __restrict__ src, _Float16* __restrict__ dst, int n) {
    int g = blockIdx.x * blockDim.x + threadIdx.x;
    if (g < n) dst[g] = (_Float16)src[g];
}

// dst[n*Kd + k] = (f16) src[k*ld + coloff + n]   (K-contiguous transposed weight panel)
__global__ void k_transp(const float* __restrict__ src, int ld, int coloff,
                         int Ncols, int Kd, _Float16* __restrict__ dst) {
    int g = blockIdx.x * blockDim.x + threadIdx.x;
    if (g < Ncols * Kd) {
        int n = g / Kd, k = g % Kd;
        dst[n * Kd + k] = (_Float16)src[k * ld + coloff + n];
    }
}

__global__ void k_cxyz(const float* __restrict__ xyz, const int* __restrict__ idxc,
                       float* __restrict__ cxyz) {
    int m = blockIdx.x * blockDim.x + threadIdx.x;
    if (m < MCTR) {
        int c = idxc[m];
        cxyz[m * 3 + 0] = xyz[c * 3 + 0];
        cxyz[m * 3 + 1] = xyz[c * 3 + 1];
        cxyz[m * 3 + 2] = xyz[c * 3 + 2];
    }
}

// gathered center features, f16 (A matrix for the q GEMM)
__global__ void k_cfeat(const float* __restrict__ feats, const int* __restrict__ idxc,
                        _Float16* __restrict__ cfeat_h) {
    int g = blockIdx.x * blockDim.x + threadIdx.x;
    if (g < MCTR * DIM) {
        int m = g >> 7, d = g & 127;
        cfeat_h[g] = (_Float16)feats[idxc[m] * DIM + d];
    }
}

// ---------------- neighbor search: wave32 per center, LDS-tiled xyz, ballot first-K ------------

#define NCH 512   // xyz points staged per chunk (6 KB LDS)

__global__ __launch_bounds__(256)
void k_neighbors(const float* __restrict__ xyz, const float* __restrict__ cxyz,
                 int* __restrict__ nbr) {
    __shared__ float sxyz[NCH * 3];
    const int t = threadIdx.x, lane = t & 31, wv = t >> 5;
    const int m = blockIdx.x * 8 + wv;            // 8 waves = 8 centers per block
    const float cx = cxyz[m * 3 + 0], cy = cxyz[m * 3 + 1], cz = cxyz[m * 3 + 2];
    int cnt = 0;
    for (int base = 0; base < NPTS; base += NCH) {
        for (int i = t; i < NCH * 3; i += 256) sxyz[i] = xyz[base * 3 + i];
        __syncthreads();
        if (cnt < KNB) {                          // wave-uniform
            for (int j0 = 0; j0 < NCH && cnt < KNB; j0 += 32) {
                int j = j0 + lane;
                float dx = sxyz[j * 3 + 0] - cx;
                float dy = sxyz[j * 3 + 1] - cy;
                float dz = sxyz[j * 3 + 2] - cz;
                bool pred = (dx * dx + dy * dy + dz * dz) < RADIUS2;
                unsigned m32 = (unsigned)__ballot(pred);
                int rank = __popc(m32 & ((1u << lane) - 1u));
                if (pred && (cnt + rank) < KNB) nbr[m * KNB + cnt + rank] = base + j;
                cnt += __popc(m32);
            }
        }
        __syncthreads();
    }
    if (cnt > KNB) cnt = KNB;
    for (int p = cnt + lane; p < KNB; p += 32) nbr[m * KNB + p] = NPTS;
}

// ---------------- nearest center per point: LDS-tiled center table ----------------

#define CCH 256   // centers staged per chunk (3 KB LDS)

__global__ __launch_bounds__(256)
void k_nearest(const float* __restrict__ xyz, const float* __restrict__ cxyz,
               int* __restrict__ nearest) {
    __shared__ float sc[CCH * 3];
    const int n = blockIdx.x * 256 + threadIdx.x;
    const float px = xyz[n * 3 + 0], py = xyz[n * 3 + 1], pz = xyz[n * 3 + 2];
    float best = 3.4e38f; int bi = 0;
    for (int base = 0; base < MCTR; base += CCH) {
        for (int i = threadIdx.x; i < CCH * 3; i += 256) sc[i] = cxyz[base * 3 + i];
        __syncthreads();
        for (int j = 0; j < CCH; ++j) {
            float dx = sc[j * 3 + 0] - px;
            float dy = sc[j * 3 + 1] - py;
            float dz = sc[j * 3 + 2] - pz;
            float d2 = dx * dx + dy * dy + dz * dz;
            if (d2 < best) { best = d2; bi = base + j; }   // strict < : first-min tiebreak
        }
        __syncthreads();
    }
    nearest[n] = bi;
}

// ---------------- generic WMMA GEMM: C[M x N] = A[M x K] @ Bt[N x K]^T (+bias, relu) ------------
// wave per 16x16 output tile; A row-major f16 (K contiguous), Bt K-contiguous f16.
// Fragments loaded straight from global (all operands L2-resident on 192MB L2).

template<int KD, bool RELU>
__global__ __launch_bounds__(256) void k_gemm(const _Float16* __restrict__ A,
                                              const _Float16* __restrict__ Bt,
                                              const float* __restrict__ bias,
                                              float* __restrict__ C,
                                              _Float16* __restrict__ Ch,
                                              int Ncols) {
    const int wid  = (blockIdx.x * blockDim.x + threadIdx.x) >> 5;
    const int lane = threadIdx.x & 31;
    const int tilesX = Ncols >> 4;
    const int mB = (wid / tilesX) << 4;
    const int nB = (wid % tilesX) << 4;
    const int rsub = lane & 15;
    const int kg   = lane >> 4;
    const _Float16* arow = A  + (size_t)(mB + rsub) * KD;
    const _Float16* bcol = Bt + (size_t)(nB + rsub) * KD;
    v8f cacc = {};
    #pragma unroll
    for (int k0 = 0; k0 < KD; k0 += 32) {
        v8h lo = *(const v8h*)(arow + k0 + kg * 16);
        v8h hi = *(const v8h*)(arow + k0 + kg * 16 + 8);
        v16h a = __builtin_shufflevector(lo, hi, 0,1,2,3,4,5,6,7,8,9,10,11,12,13,14,15);
        v16h b = *(const v16h*)(bcol + k0 + kg * 16);
        cacc = __builtin_amdgcn_wmma_f32_16x16x32_f16(false, a, false, b, (short)0, cacc,
                                                      false, false);
    }
    const int n  = nB + rsub;
    const int m0 = mB + 8 * kg;
    const float bv = bias ? bias[n] : 0.f;
    #pragma unroll
    for (int r = 0; r < 8; ++r) {
        float v = cacc[r] + bv;
        if (RELU) v = v > 0.f ? v : 0.f;
        if (C)  C [(size_t)(m0 + r) * Ncols + n] = v;
        if (Ch) Ch[(size_t)(m0 + r) * Ncols + n] = (_Float16)v;
    }
}

// ---------------- per-center attention: async gather -> WMMA KV GEMM -> wave32 softmax ---------

__global__ __launch_bounds__(256)
void k_attn(const _Float16* __restrict__ feats_h, const int* __restrict__ nbr,
            const _Float16* __restrict__ WkvT, const float* __restrict__ qbuf,
            _Float16* __restrict__ out_h) {
    __shared__ __attribute__((aligned(16))) _Float16 sA[32 * AS];
    __shared__ float sKV[32 * KVS];   // cols 0..127 = K, 128..255 = V
    __shared__ float sQ[DIM];
    __shared__ float sP[DIM];         // 4 heads x 32 attention probs

    const int m    = blockIdx.x;
    const int t    = threadIdx.x;
    const int lane = t & 31;
    const int wv   = t >> 5;

    __builtin_prefetch(&WkvT[(t & 255) * DIM], 0, 0);   // global_prefetch_b8 of the B panel

    if (t < DIM) sQ[t] = qbuf[m * DIM + t];

#if USE_ASYNC_LDS
    // 32 rows x 256B, moved as 512 async B128 transfers straight into LDS (no VGPR round-trip)
    for (int e = t; e < 32 * 16; e += 256) {
        int row = e >> 4, ch = e & 15;
        int idx = nbr[m * KNB + row];
        int safe = idx < NPTS ? idx : (NPTS - 1);
        const _Float16* gsrc = feats_h + (size_t)safe * DIM + ch * 8;
        __builtin_amdgcn_global_load_async_to_lds_b128(
            ASYNC_GPTR(gsrc), ASYNC_SPTR(&sA[row * AS + ch * 8]), 0, 0);
    }
#if __has_builtin(__builtin_amdgcn_s_wait_asynccnt)
    __builtin_amdgcn_s_wait_asynccnt(0);
#else
    asm volatile("s_wait_asynccnt 0x0" ::: "memory");
#endif
#else
    for (int e = t; e < 32 * DIM; e += 256) {
        int row = e >> 7, col = e & 127;
        int idx = nbr[m * KNB + row];
        int safe = idx < NPTS ? idx : (NPTS - 1);
        sA[row * AS + col] = feats_h[safe * DIM + col];
    }
#endif
    __syncthreads();

    // KV = A(32x128 f16) @ WkvT(128x256)  -> sKV(32x256 f32)
    {
        const int rsub = lane & 15;
        const int kg   = lane >> 4;
        for (int tile = wv; tile < 32; tile += 8) {
            const int mB = (tile >> 4) * 16;
            const int nB = (tile & 15) * 16;
            const int row  = mB + rsub;
            const int ncol = nB + rsub;
            v8f cacc = {};
            #pragma unroll
            for (int ks = 0; ks < 4; ++ks) {
                const int k0 = ks * 32;
                v8h lo = *(const v8h*)&sA[row * AS + k0 + kg * 16];
                v8h hi = *(const v8h*)&sA[row * AS + k0 + kg * 16 + 8];
                v16h a = __builtin_shufflevector(lo, hi, 0,1,2,3,4,5,6,7,8,9,10,11,12,13,14,15);
                v16h b = *(const v16h*)&WkvT[ncol * DIM + k0 + kg * 16];
                cacc = __builtin_amdgcn_wmma_f32_16x16x32_f16(false, a, false, b, (short)0,
                                                              cacc, false, false);
            }
            const int m0 = mB + 8 * kg;
            #pragma unroll
            for (int r = 0; r < 8; ++r) sKV[(m0 + r) * KVS + ncol] = cacc[r];
        }
    }
    __syncthreads();

    // scores + per-head wave32 softmax (waves 0..3 = heads, lane = neighbor)
    if (t < DIM) {
        const int h = t >> 5, kk = lane;
        const float* qh   = &sQ[h * HDIM];
        const float* krow = &sKV[kk * KVS + h * HDIM];
        float s = 0.f;
        #pragma unroll 8
        for (int d = 0; d < HDIM; ++d) s += qh[d] * krow[d];
        s *= ATT_SCALE;
        if (nbr[m * KNB + kk] >= NPTS) s = -1e30f;
        float mx = s;
        for (int off = 16; off; off >>= 1) mx = fmaxf(mx, __shfl_xor(mx, off));
        float e = __expf(s - mx);
        float sum = e;
        for (int off = 16; off; off >>= 1) sum += __shfl_xor(sum, off);
        sP[t] = e / sum;
    }
    __syncthreads();

    // out = attn @ V ; emit f16 row for the Wout WMMA GEMM
    if (t < DIM) {
        const int h = t >> 5, d = lane;
        const float* ph = &sP[h * KNB];
        float acc = 0.f;
        #pragma unroll 8
        for (int kk = 0; kk < KNB; ++kk) acc += ph[kk] * sKV[kk * KVS + 128 + h * HDIM + d];
        out_h[m * DIM + t] = (_Float16)acc;
    }
}

// ---------------- LN + residual:  y = resid + LN(x) ----------------
// if idxc != null: resid row = base[idxc[m]] (gather), else base[m]

__global__ __launch_bounds__(128)
void k_ln_res(const float* __restrict__ base, const int* __restrict__ idxc,
              const float* __restrict__ xin,
              const float* __restrict__ ln_w, const float* __restrict__ ln_b,
              float* __restrict__ yout, _Float16* __restrict__ yh) {
    __shared__ float sR[DIM];
    __shared__ float sMV[2];
    const int m = blockIdx.x, t = threadIdx.x, lane = t & 31;
    float x = xin[m * DIM + t];
    sR[t] = x;
    __syncthreads();
    if (t < 32) {
        float s1 = 0.f, s2 = 0.f;
        for (int i = lane; i < DIM; i += 32) { float v = sR[i]; s1 += v; s2 += v * v; }
        for (int off = 16; off; off >>= 1) { s1 += __shfl_xor(s1, off); s2 += __shfl_xor(s2, off); }
        if (lane == 0) {
            float mu = s1 * (1.f / DIM);
            float var = s2 * (1.f / DIM) - mu * mu;
            sMV[0] = mu; sMV[1] = rsqrtf(var + LN_EPS);
        }
    }
    __syncthreads();
    int row = idxc ? idxc[m] : m;
    float y = base[row * DIM + t] + (x - sMV[0]) * sMV[1] * ln_w[t] + ln_b[t];
    yout[m * DIM + t] = y;
    if (yh) yh[m * DIM + t] = (_Float16)y;
}

// ---------------- upsample: out = feats + cf[nearest] ----------------

__global__ void k_upsample(const float* __restrict__ feats, const float* __restrict__ cf,
                           const int* __restrict__ nearest, float* __restrict__ out) {
    int g = blockIdx.x * blockDim.x + threadIdx.x;
    if (g >= NPTS * DIM) return;
    int n = g >> 7, d = g & 127;
    out[g] = feats[g] + cf[nearest[n] * DIM + d];
}

// ---------------- launcher ----------------

extern "C" void kernel_launch(void* const* d_in, const int* in_sizes, int n_in,
                              void* d_out, int out_size, void* d_ws, size_t ws_size,
                              hipStream_t stream) {
    const float* xyz   = (const float*)d_in[0];
    const float* feats = (const float*)d_in[1];
    const int*   idxc  = (const int*)  d_in[2];
    const float* Wqkv  = (const float*)d_in[3];
    const float* Wout  = (const float*)d_in[4];
    const float* b_out = (const float*)d_in[5];
    const float* ln_w  = (const float*)d_in[6];
    const float* ln_b  = (const float*)d_in[7];
    const float* W1    = (const float*)d_in[8];
    const float* b1    = (const float*)d_in[9];
    const float* W2    = (const float*)d_in[10];
    const float* b2    = (const float*)d_in[11];
    float* out = (float*)d_out;

    char* ws = (char*)d_ws;
    size_t off = 0;
    auto alloc = [&](size_t bytes) -> void* {
        void* p = ws + off;
        off = (off + bytes + 255) & ~(size_t)255;
        return p;
    };
    _Float16* feats_h = (_Float16*)alloc((size_t)NPTS * DIM * 2);
    _Float16* WqT     = (_Float16*)alloc((size_t)128 * 128 * 2);
    _Float16* WkvT    = (_Float16*)alloc((size_t)256 * 128 * 2);
    _Float16* WoutT   = (_Float16*)alloc((size_t)128 * 128 * 2);
    _Float16* W1T     = (_Float16*)alloc((size_t)512 * 128 * 2);
    _Float16* W2T     = (_Float16*)alloc((size_t)128 * 512 * 2);
    float*    cxyz    = (float*)   alloc((size_t)MCTR * 3 * 4);
    int*      nbr     = (int*)     alloc((size_t)MCTR * KNB * 4);
    int*      nearest = (int*)     alloc((size_t)NPTS * 4);
    _Float16* cfeat_h = (_Float16*)alloc((size_t)MCTR * DIM * 2);
    float*    qbuf    = (float*)   alloc((size_t)MCTR * DIM * 4);
    _Float16* out_h   = (_Float16*)alloc((size_t)MCTR * DIM * 2);
    float*    upd     = (float*)   alloc((size_t)MCTR * DIM * 4);
    float*    cf1     = (float*)   alloc((size_t)MCTR * DIM * 4);
    _Float16* cf1_h   = (_Float16*)alloc((size_t)MCTR * DIM * 2);
    _Float16* hid_h   = (_Float16*)alloc((size_t)MCTR * 512 * 2);
    float*    ybuf    = (float*)   alloc((size_t)MCTR * DIM * 4);
    float*    cf      = (float*)   alloc((size_t)MCTR * DIM * 4);
    (void)in_sizes; (void)n_in; (void)out_size; (void)ws_size;

    // conversions / transposed weight panels
    k_f2h<<<(NPTS * DIM + 255) / 256, 256, 0, stream>>>(feats, feats_h, NPTS * DIM);
    k_transp<<<(128 * 128 + 255) / 256, 256, 0, stream>>>(Wqkv, 384,   0, 128, 128, WqT);
    k_transp<<<(256 * 128 + 255) / 256, 256, 0, stream>>>(Wqkv, 384, 128, 256, 128, WkvT);
    k_transp<<<(128 * 128 + 255) / 256, 256, 0, stream>>>(Wout, 128,   0, 128, 128, WoutT);
    k_transp<<<(512 * 128 + 255) / 256, 256, 0, stream>>>(W1,   512,   0, 512, 128, W1T);
    k_transp<<<(128 * 512 + 255) / 256, 256, 0, stream>>>(W2,   128,   0, 128, 512, W2T);
    k_cxyz <<<(MCTR + 255) / 256, 256, 0, stream>>>(xyz, idxc, cxyz);
    k_cfeat<<<(MCTR * DIM + 255) / 256, 256, 0, stream>>>(feats, idxc, cfeat_h);

    // geometry (LDS-tiled scans)
    k_neighbors<<<MCTR / 8, 256, 0, stream>>>(xyz, cxyz, nbr);
    k_nearest<<<NPTS / 256, 256, 0, stream>>>(xyz, cxyz, nearest);

    // q = center_feat @ Wq           (4096x128x128, WMMA)
    k_gemm<128, false><<<(MCTR / 16) * (128 / 16) / 8, 256, 0, stream>>>(
        cfeat_h, WqT, nullptr, qbuf, nullptr, 128);

    // attention (async gather + gathered-KV WMMA + softmax)
    k_attn<<<MCTR, 256, 0, stream>>>(feats_h, nbr, WkvT, qbuf, out_h);

    // updated = out @ Wout + b_out   (4096x128x128, WMMA)
    k_gemm<128, false><<<(MCTR / 16) * (128 / 16) / 8, 256, 0, stream>>>(
        out_h, WoutT, b_out, upd, nullptr, 128);

    // cf1 = center_feat + LN(updated)
    k_ln_res<<<MCTR, 128, 0, stream>>>(feats, idxc, upd, ln_w, ln_b, cf1, cf1_h);

    // hid = relu(cf1 @ W1 + b1)      (4096x512x128, WMMA)
    k_gemm<128, true><<<(MCTR / 16) * (512 / 16) / 8, 256, 0, stream>>>(
        cf1_h, W1T, b1, nullptr, hid_h, 512);

    // y = hid @ W2 + b2              (4096x128x512, WMMA)
    k_gemm<512, false><<<(MCTR / 16) * (128 / 16) / 8, 256, 0, stream>>>(
        hid_h, W2T, b2, ybuf, nullptr, 128);

    // cf = cf1 + LN(y)
    k_ln_res<<<MCTR, 128, 0, stream>>>(cf1, nullptr, ybuf, ln_w, ln_b, cf, nullptr);

    // out = feats + cf[nearest]
    k_upsample<<<(NPTS * DIM + 255) / 256, 256, 0, stream>>>(feats, cf, nearest, out);
}
